// PointNet2Classification_50568944943632
// MI455X (gfx1250) — matrix operand order repaired and checked by reading, not compiled, not run on hardware
//
#include <hip/hip_runtime.h>
#include <hip/hip_bf16.h>
#include <cstdint>
#include <cstddef>

typedef _Float16 f16;
typedef __attribute__((ext_vector_type(16))) _Float16 v16h;
typedef __attribute__((ext_vector_type(8)))  _Float16 v8h;
typedef __attribute__((ext_vector_type(8)))  float    v8f;

// ---------------------------------------------------------------------------
// Input split/transpose: (B,6,N) -> xyz (B,N,3), norm (B,N,3)
// ---------------------------------------------------------------------------
__global__ void split_transpose_kernel(const float* __restrict__ in,
                                       float* __restrict__ xyzT,
                                       float* __restrict__ normT,
                                       int Bn, int Nn) {
    int t = blockIdx.x * 256 + threadIdx.x;
    int total = Bn * 6 * Nn;
    if (t >= total) return;
    int b  = t / (6 * Nn);
    int r  = t - b * 6 * Nn;
    int ch = r / Nn;
    int nn = r - ch * Nn;
    float v = in[t];
    if (ch < 3) xyzT [((size_t)b * Nn + nn) * 3 + ch]       = v;
    else        normT[((size_t)b * Nn + nn) * 3 + (ch - 3)] = v;
}

// ---------------------------------------------------------------------------
// Farthest point sampling: one block per batch, dist kept in registers.
// Tie-break: first (lowest) index of the max, matching jnp.argmax.
// ---------------------------------------------------------------------------
__global__ __launch_bounds__(256)
void fps_kernel(const float* __restrict__ xyz, int n, int npoint,
                int* __restrict__ idx, float* __restrict__ new_xyz) {
    int b = blockIdx.x, tid = threadIdx.x;
    const float* base = xyz + (size_t)b * n * 3;
    float dist[16];
    int chunks = (n + 255) >> 8;
    for (int i = 0; i < chunks; ++i) dist[i] = 1e10f;
    __shared__ float sv[256];
    __shared__ int   si[256];
    __shared__ int   sfar;
    if (tid == 0) sfar = 0;
    __syncthreads();
    for (int p = 0; p < npoint; ++p) {
        int far = sfar;
        if (tid == 0) idx[b * npoint + p] = far;
        float cx = base[far * 3 + 0], cy = base[far * 3 + 1], cz = base[far * 3 + 2];
        float bv = -1.0f; int bi = 0;
        for (int i = 0; i < chunks; ++i) {
            int j = tid + (i << 8);
            if (j < n) {
                float dx = base[j * 3 + 0] - cx;
                float dy = base[j * 3 + 1] - cy;
                float dz = base[j * 3 + 2] - cz;
                float d  = dx * dx + dy * dy + dz * dz;
                float nd = fminf(dist[i], d);
                dist[i] = nd;
                if (nd > bv) { bv = nd; bi = j; }   // increasing j => first max kept
            }
        }
        sv[tid] = bv; si[tid] = bi;
        __syncthreads();
        for (int s = 128; s > 0; s >>= 1) {
            if (tid < s) {
                if (sv[tid + s] > sv[tid] ||
                    (sv[tid + s] == sv[tid] && si[tid + s] < si[tid])) {
                    sv[tid] = sv[tid + s]; si[tid] = si[tid + s];
                }
            }
            __syncthreads();
        }
        if (tid == 0) sfar = si[0];
        __syncthreads();
    }
    for (int t = tid; t < npoint * 3; t += 256) {
        int p = t / 3, k = t - p * 3;
        new_xyz[((size_t)b * npoint + p) * 3 + k] = base[(size_t)idx[b * npoint + p] * 3 + k];
    }
}

// ---------------------------------------------------------------------------
// Ball query: first K indices (ascending) with d2 <= r2; pad with first hit.
// ---------------------------------------------------------------------------
__global__ void ball_query_kernel(const float* __restrict__ xyz,
                                  const float* __restrict__ new_xyz,
                                  int n, int S, float r2, int K,
                                  int* __restrict__ gi, int total) {
    int t = blockIdx.x * 128 + threadIdx.x;
    if (t >= total) return;            // t = b*S + s
    int b = t / S;
    const float* base = xyz + (size_t)b * n * 3;
    float cx = new_xyz[t * 3 + 0], cy = new_xyz[t * 3 + 1], cz = new_xyz[t * 3 + 2];
    int* out = gi + (size_t)t * K;
    int cnt = 0;
    for (int j = 0; j < n && cnt < K; ++j) {
        float dx = base[j * 3 + 0] - cx;
        float dy = base[j * 3 + 1] - cy;
        float dz = base[j * 3 + 2] - cz;
        if (dx * dx + dy * dy + dz * dz <= r2) out[cnt++] = j;
    }
    if (cnt == 0) { out[0] = 0; cnt = 1; }
    int f = out[0];
    for (; cnt < K; ++cnt) out[cnt] = f;
}

// ---------------------------------------------------------------------------
// Group/gather (thread-per-element): row c-layout [pts(Cp), xyz-ctr(3), pad]
// ---------------------------------------------------------------------------
__global__ void group_kernel(const float* __restrict__ xyz,
                             const float* __restrict__ pts,
                             const float* __restrict__ new_xyz,
                             const int* __restrict__ gi,
                             int n, int S, int K, int Cp, int Cpad,
                             f16* __restrict__ gp, long long total) {
    long long t = (long long)blockIdx.x * 256 + threadIdx.x;
    if (t >= total) return;
    int  c   = (int)(t % Cpad);
    long long row = t / Cpad;          // 0 .. B*S*K-1
    int  bs  = (int)(row / K);
    int  b   = bs / S;
    int  id  = gi[row];
    float v = 0.0f;
    if (c < Cp) {
        v = pts[((size_t)b * n + id) * Cp + c];
    } else if (c < Cp + 3) {
        int d = c - Cp;
        v = xyz[((size_t)b * n + id) * 3 + d] - new_xyz[(size_t)bs * 3 + d];
    }
    gp[t] = (f16)v;
}

// sa3 grouping: [l2_xyz(3), l2_pts(640), pad -> 672] for 16*128 rows
__global__ void concat3_kernel(const float* __restrict__ l2xyz,
                               const float* __restrict__ l2pts,
                               f16* __restrict__ gp) {
    int row = blockIdx.x;              // 0..2047 = b*128 + k
    f16* out = gp + (size_t)row * 672;
    for (int c = threadIdx.x; c < 672; c += blockDim.x) {
        float v = 0.0f;
        if (c < 3)        v = l2xyz[row * 3 + c];
        else if (c < 643) v = l2pts[(size_t)row * 640 + (c - 3)];
        out[c] = (f16)v;
    }
}

// ---------------------------------------------------------------------------
// Weight pad + convert: W[O,C] f32 -> Wpad[Opad,Cpad] f16 (zero padded)
// ---------------------------------------------------------------------------
__global__ void wpad_kernel(const float* __restrict__ w, f16* __restrict__ out,
                            int O, int C, int Cpad, int Opad, int total) {
    int t = blockIdx.x * 256 + threadIdx.x;
    if (t >= total) return;
    int o = t / Cpad, c = t - o * Cpad;
    out[t] = (o < O && c < C) ? (f16)w[(size_t)o * C + c] : (f16)0.0f;
}

// ---------------------------------------------------------------------------
// WMMA GEMM: Z[M,O] (f16) = A[M,Cpad] x W[O,Cpad]^T, f32 accumulate.
// Each wave: one 16-row tile x NT 16-col tiles; A fragment reused by NT WMMAs.
// Ping-pong software pipeline, 2x unrolled so the buffer rotation needs NO
// register-register copies: each fragment set (X/Y) is only ever rewritten by
// loads. WMMAs for step i-1 overlap the in-flight loads of step i, retiring
// on partial loadcnt waits. A/B fragments use contiguous k = 16*(lane/16) + s (wave32).
// ---------------------------------------------------------------------------
template <int NT>
__global__ __launch_bounds__(128)
void gemm_f16_kernel(const f16* __restrict__ A, const f16* __restrict__ W,
                     f16* __restrict__ Z, int M, int Cpad, int O) {
    int lane = threadIdx.x;
    int mt = blockIdx.x * 4 + threadIdx.y;
    if (mt * 16 >= M) return;
    int m0 = mt * 16;
    int o0 = blockIdx.y * (16 * NT);

    int mrow = m0 + (lane & 15);
    int agrp = (lane >> 4) << 3;                 // 0 or 8 (A K-subgroup)
    const f16* arow = A + (size_t)mrow * Cpad;

    int kb = (lane >> 4) << 4;                   // 0 or 16 (B K-half)
    const f16* bbase = W + (size_t)(o0 + (lane & 15)) * Cpad + kb;
    const size_t bstride = (size_t)16 * Cpad;    // next 16-col tile of W

    v8f acc[NT];
#pragma unroll
    for (int t = 0; t < NT; ++t) acc[t] = (v8f){};

    auto loadStep = [&](int k0, v16h& a, v16h (&bb)[NT]) {
        v8h alo = *(const v8h*)(arow + k0 + kb);          // K = k0+kb..+7
        v8h ahi = *(const v8h*)(arow + k0 + kb + 8);      // K = k0+kb+8..+15
#pragma unroll
        for (int i = 0; i < 8; ++i) { a[i] = alo[i]; a[8 + i] = ahi[i]; }
#pragma unroll
        for (int t = 0; t < NT; ++t)
            bb[t] = *(const v16h*)(bbase + (size_t)t * bstride + k0);
    };
    auto compute = [&](const v16h& a, const v16h (&bb)[NT]) {
#pragma unroll
        for (int t = 0; t < NT; ++t)
            acc[t] = __builtin_amdgcn_wmma_f32_16x16x32_f16(
                false, a, false, bb[t], (short)0, acc[t], false, false);
    };

    v16h aX, aY;
    v16h bX[NT], bY[NT];
    int KS = Cpad >> 5;                 // number of 32-wide k-steps (>= 1)
    loadStep(0, aX, bX);                // step 0 -> X
    int i = 1;
    // invariant at loop top: X holds step i-1 (unconsumed), steps < i-1 done
    for (; i + 1 < KS; i += 2) {
        __builtin_prefetch(arow + (i << 5) + 256, 0, 1);
        loadStep(i << 5, aY, bY);       // step i   -> Y (in flight)
        compute(aX, bX);                // step i-1
        loadStep((i + 1) << 5, aX, bX); // step i+1 -> X (in flight)
        compute(aY, bY);                // step i
    }
    if (i < KS) {                       // one step left to load (i == KS-1)
        loadStep(i << 5, aY, bY);
        compute(aX, bX);                // step i-1
        compute(aY, bY);                // step i (last)
    } else {                            // all loaded (i == KS)
        compute(aX, bX);                // last step
    }

    int mbase = m0 + ((lane >> 4) << 3);
#pragma unroll
    for (int t = 0; t < NT; ++t) {
        f16* zc = Z + o0 + t * 16 + (lane & 15);
#pragma unroll
        for (int r = 0; r < 8; ++r)
            zc[(size_t)(mbase + r) * O] = (f16)acc[t][r];
    }
}

// ---------------------------------------------------------------------------
// BatchNorm stats, pass 1: coalesced row-range scan, LDS per-channel
// accumulation (ds_add_f32), one global atomic per channel per block.
// ---------------------------------------------------------------------------
__global__ void zero_f32_kernel(float* __restrict__ p, int n) {
    int t = blockIdx.x * 256 + threadIdx.x;
    if (t < n) p[t] = 0.0f;
}

__global__ __launch_bounds__(256)
void bn_partial_kernel(const f16* __restrict__ Z, long long total, int O,
                       int rows_per_block,
                       float* __restrict__ gsum, float* __restrict__ gsum2) {
    extern __shared__ float sh[];        // [0..O) sum, [O..2O) sumsq
    float* s1 = sh;
    float* s2 = sh + O;
    for (int c = threadIdx.x; c < 2 * O; c += 256) sh[c] = 0.0f;
    __syncthreads();
    long long base  = (long long)blockIdx.x * rows_per_block * O;
    int       count = rows_per_block * O;
    for (int i = threadIdx.x; i < count; i += 256) {
        long long t = base + i;
        if (t < total) {
            float v = (float)Z[t];       // contiguous, coalesced
            int o = i % O;
            atomicAdd(&s1[o], v);
            atomicAdd(&s2[o], v * v);
        }
    }
    __syncthreads();
    for (int o = threadIdx.x; o < O; o += 256) {
        atomicAdd(&gsum[o],  s1[o]);
        atomicAdd(&gsum2[o], s2[o]);
    }
}

__global__ void bn_finalize_kernel(const float* __restrict__ gsum,
                                   const float* __restrict__ gsum2,
                                   int O, float Minv,
                                   float* __restrict__ mean,
                                   float* __restrict__ rsig) {
    int o = blockIdx.x * 256 + threadIdx.x;
    if (o >= O) return;
    float mu  = gsum[o] * Minv;
    float var = gsum2[o] * Minv - mu * mu;
    mean[o] = mu;
    rsig[o] = rsqrtf(var + 1e-5f);
}

__global__ void bn_relu_kernel(const f16* __restrict__ Z,
                               const float* __restrict__ mean,
                               const float* __restrict__ rsig,
                               const float* __restrict__ g,
                               const float* __restrict__ beta,
                               long long total, int O, f16* __restrict__ out) {
    long long t = (long long)blockIdx.x * 256 + threadIdx.x;
    if (t >= total) return;
    int o = (int)(t % O);
    float v = ((float)Z[t] - mean[o]) * rsig[o] * g[o] + beta[o];
    out[t] = (f16)fmaxf(v, 0.0f);
}

// ---------------------------------------------------------------------------
// Max pool over K neighbors -> f32 into concatenated output at channel offset
// ---------------------------------------------------------------------------
__global__ void maxpool_kernel(const f16* __restrict__ Y, int BS, int K, int O,
                               float* __restrict__ out, int Ctot, int coff,
                               int total) {
    int t = blockIdx.x * 256 + threadIdx.x;
    if (t >= total) return;               // t = bs*O + o
    int bs = t / O, o = t - bs * O;
    const f16* p = Y + ((size_t)bs * K) * O + o;
    float m = -3.0e38f;
    for (int k = 0; k < K; ++k) m = fmaxf(m, (float)p[(size_t)k * O]);
    out[(size_t)bs * Ctot + coff + o] = m;
}

__global__ void f32_to_f16_kernel(const float* __restrict__ in, f16* __restrict__ out,
                                  int total) {
    int t = blockIdx.x * 256 + threadIdx.x;
    if (t < total) out[t] = (f16)in[t];
}

__global__ void copy_f32_kernel(const float* __restrict__ in, float* __restrict__ out,
                                int total) {
    int t = blockIdx.x * 256 + threadIdx.x;
    if (t < total) out[t] = in[t];
}

// fc3: logits = Z[b, 0..39] + bias, then log_softmax into d_out[b*40 + c]
__global__ __launch_bounds__(64)
void fc3_logsoftmax_kernel(const f16* __restrict__ Z, const float* __restrict__ bias,
                           float* __restrict__ out) {
    int b = blockIdx.x, tid = threadIdx.x;
    __shared__ float sh[64];
    float z = (tid < 40) ? (float)Z[b * 48 + tid] + bias[tid] : -3.0e38f;
    sh[tid] = z;
    __syncthreads();
    for (int s = 32; s > 0; s >>= 1) {
        if (tid < s) sh[tid] = fmaxf(sh[tid], sh[tid + s]);
        __syncthreads();
    }
    float mx = sh[0];
    __syncthreads();
    sh[tid] = (tid < 40) ? expf(z - mx) : 0.0f;
    __syncthreads();
    for (int s = 32; s > 0; s >>= 1) {
        if (tid < s) sh[tid] += sh[tid + s];
        __syncthreads();
    }
    float lse = logf(sh[0]);
    if (tid < 40) out[b * 40 + tid] = z - mx - lse;
}

// ---------------------------------------------------------------------------
// Host orchestration
// ---------------------------------------------------------------------------
extern "C" void kernel_launch(void* const* d_in, const int* in_sizes, int n_in,
                              void* d_out, int out_size, void* d_ws, size_t ws_size,
                              hipStream_t stream) {
    (void)in_sizes; (void)out_size; (void)ws_size;
    const int B = 16, N = 4096;
    const float* xin = (const float*)d_in[0];

    // ---- parameter walk (setup_inputs insertion order; w,b,g,beta per layer)
    int pi = 1;
    auto nf = [&]() -> const float* {
        int i = (pi < n_in) ? pi : (n_in - 1);
        ++pi;
        return (const float*)d_in[i];
    };
    struct Layer { const float *w, *b, *g, *beta; int cin, cout; };
    const int sa1_outs[3][3] = {{32,32,64},{64,64,128},{64,96,128}};
    const int sa2_outs[3][3] = {{64,64,128},{128,128,256},{128,128,256}};
    const int sa3_outs[3]    = {256,512,1024};
    Layer sa1[3][3], sa2[3][3], sa3l[3];
    for (int blk = 0; blk < 3; ++blk) {
        int cin = 6;
        for (int l = 0; l < 3; ++l) {
            sa1[blk][l] = {nf(), nf(), nf(), nf(), cin, sa1_outs[blk][l]};
            cin = sa1_outs[blk][l];
        }
    }
    for (int blk = 0; blk < 3; ++blk) {
        int cin = 323;
        for (int l = 0; l < 3; ++l) {
            sa2[blk][l] = {nf(), nf(), nf(), nf(), cin, sa2_outs[blk][l]};
            cin = sa2_outs[blk][l];
        }
    }
    {
        int cin = 643;
        for (int l = 0; l < 3; ++l) {
            sa3l[l] = {nf(), nf(), nf(), nf(), cin, sa3_outs[l]};
            cin = sa3_outs[l];
        }
    }
    Layer fc1 = {nf(), nf(), nf(), nf(), 1024, 512};
    Layer fc2 = {nf(), nf(), nf(), nf(),  512, 256};
    const float* fc3w = nf();
    const float* fc3b = nf();

    // ---- workspace bump allocator
    uint8_t* wsb = (uint8_t*)d_ws;
    size_t off = 0;
    auto alloc = [&](size_t bytes) -> void* {
        void* r = wsb + off;
        off = (off + bytes + 255) & ~(size_t)255;
        return r;
    };
    float* xyzT     = (float*)alloc((size_t)B * N * 3 * 4);
    float* normT    = (float*)alloc((size_t)B * N * 3 * 4);
    int*   fps1_idx = (int*)  alloc((size_t)B * 512 * 4);
    float* new_xyz1 = (float*)alloc((size_t)B * 512 * 3 * 4);
    float* l1_pts   = (float*)alloc((size_t)B * 512 * 320 * 4);
    int*   fps2_idx = (int*)  alloc((size_t)B * 128 * 4);
    float* new_xyz2 = (float*)alloc((size_t)B * 128 * 3 * 4);
    float* l2_pts   = (float*)alloc((size_t)B * 128 * 640 * 4);
    float* l3       = (float*)alloc((size_t)B * 1024 * 4);
    int*   gi       = (int*)  alloc((size_t)B * 512 * 128 * 4);
    f16*   wpad     = (f16*)  alloc((size_t)1048576 * 2);
    float* mean     = (float*)alloc(4096);
    float* rsig     = (float*)alloc(4096);
    float* gsum     = (float*)alloc(4096);
    float* gsum2    = (float*)alloc(4096);
    const size_t BIGN = 134217728;           // max M*C elements (1M x 128)
    f16* bufA = (f16*)alloc(BIGN * 2);
    f16* bufB = (f16*)alloc(BIGN * 2);

    // ---- GEMM launcher: pick widest per-wave o-tiling O allows
    auto launch_gemm = [&](const f16* Ain, const f16* Wp, f16* Zout,
                           int M, int Cpad, int O) {
        dim3 t(32, 4);
        unsigned mg = (unsigned)((M / 16 + 3) / 4);
        if (O % 64 == 0) {
            dim3 g(mg, O / 64);
            gemm_f16_kernel<4><<<g, t, 0, stream>>>(Ain, Wp, Zout, M, Cpad, O);
        } else if (O % 32 == 0) {
            dim3 g(mg, O / 32);
            gemm_f16_kernel<2><<<g, t, 0, stream>>>(Ain, Wp, Zout, M, Cpad, O);
        } else {
            dim3 g(mg, O / 16);
            gemm_f16_kernel<1><<<g, t, 0, stream>>>(Ain, Wp, Zout, M, Cpad, O);
        }
    };

    // ---- one MLP layer: pad weights, WMMA GEMM, BN stats, BN+ReLU
    auto run_layer = [&](const Layer& L, int M, int Cpad) {
        int O = L.cout;
        int wtot = O * Cpad;
        wpad_kernel<<<(wtot + 255) / 256, 256, 0, stream>>>(L.w, wpad, O, L.cin, Cpad, O, wtot);
        launch_gemm(bufA, wpad, bufB, M, Cpad, O);
        long long mo = (long long)M * O;
        zero_f32_kernel<<<(O + 255) / 256, 256, 0, stream>>>(gsum, O);
        zero_f32_kernel<<<(O + 255) / 256, 256, 0, stream>>>(gsum2, O);
        int rpb = 32768 / O; if (rpb < 1) rpb = 1;
        unsigned nblk = (unsigned)((M + rpb - 1) / rpb);
        bn_partial_kernel<<<nblk, 256, (size_t)2 * O * 4, stream>>>(
            bufB, mo, O, rpb, gsum, gsum2);
        bn_finalize_kernel<<<(O + 255) / 256, 256, 0, stream>>>(
            gsum, gsum2, O, 1.0f / (float)M, mean, rsig);
        bn_relu_kernel<<<(unsigned)((mo + 255) / 256), 256, 0, stream>>>(
            bufB, mean, rsig, L.g, L.beta, mo, O, bufA);
    };

    // ---- one SA scale: ball query, group, 3 layers, maxpool
    auto run_sa_scale = [&](const float* xyzS, const float* ptsS, const float* nxyz,
                            int n, int S, int Cp, float radius, int K,
                            const Layer* Ls, float* pooled, int Ctot, int coff) {
        int BS = B * S;
        ball_query_kernel<<<(BS + 127) / 128, 128, 0, stream>>>(
            xyzS, nxyz, n, S, radius * radius, K, gi, BS);
        int M = BS * K;
        int Cpad = ((Cp + 3) + 31) / 32 * 32;
        long long gtot = (long long)M * Cpad;
        group_kernel<<<(unsigned)((gtot + 255) / 256), 256, 0, stream>>>(
            xyzS, ptsS, nxyz, gi, n, S, K, Cp, Cpad, bufA, gtot);
        int cp = Cpad;
        for (int l = 0; l < 3; ++l) { run_layer(Ls[l], M, cp); cp = Ls[l].cout; }
        int O = Ls[2].cout;
        int tot = BS * O;
        maxpool_kernel<<<(tot + 255) / 256, 256, 0, stream>>>(
            bufA, BS, K, O, pooled, Ctot, coff, tot);
    };

    // ================= pipeline =================
    {
        int tot = B * 6 * N;
        split_transpose_kernel<<<(tot + 255) / 256, 256, 0, stream>>>(xin, xyzT, normT, B, N);
    }
    // --- SA1 (npoint=512)
    fps_kernel<<<B, 256, 0, stream>>>(xyzT, N, 512, fps1_idx, new_xyz1);
    {
        const float r[3] = {0.1f, 0.2f, 0.4f};
        const int   k[3] = {16, 32, 128};
        const int   co[3] = {0, 64, 192};
        for (int s = 0; s < 3; ++s)
            run_sa_scale(xyzT, normT, new_xyz1, N, 512, 3, r[s], k[s],
                         sa1[s], l1_pts, 320, co[s]);
    }
    // --- SA2 (npoint=128)
    fps_kernel<<<B, 256, 0, stream>>>(new_xyz1, 512, 128, fps2_idx, new_xyz2);
    {
        const float r[3] = {0.2f, 0.4f, 0.8f};
        const int   k[3] = {32, 64, 128};
        const int   co[3] = {0, 128, 384};
        for (int s = 0; s < 3; ++s)
            run_sa_scale(new_xyz1, l1_pts, new_xyz2, 512, 128, 320, r[s], k[s],
                         sa2[s], l2_pts, 640, co[s]);
    }
    // --- SA3 (group-all, K=128, Cin=643 -> 672)
    {
        concat3_kernel<<<B * 128, 128, 0, stream>>>(new_xyz2, l2_pts, bufA);
        int M = B * 128, cp = 672;
        for (int l = 0; l < 3; ++l) { run_layer(sa3l[l], M, cp); cp = sa3_outs[l]; }
        int tot = B * 1024;
        maxpool_kernel<<<(tot + 255) / 256, 256, 0, stream>>>(
            bufA, B, 128, 1024, l3, 1024, 0, tot);
    }
    // l3 -> d_out[640:]
    copy_f32_kernel<<<(B * 1024 + 255) / 256, 256, 0, stream>>>(
        l3, (float*)d_out + B * 40, B * 1024);
    // --- FC head
    f32_to_f16_kernel<<<(B * 1024 + 255) / 256, 256, 0, stream>>>(l3, bufA, B * 1024);
    run_layer(fc1, B, 1024);
    run_layer(fc2, B, 512);
    {   // fc3: O=40 padded to 48, bias added in log-softmax kernel
        int wtot = 48 * 256;
        wpad_kernel<<<(wtot + 255) / 256, 256, 0, stream>>>(fc3w, wpad, 40, 256, 256, 48, wtot);
        dim3 g(1, 3), t(32, 4);
        gemm_f16_kernel<1><<<g, t, 0, stream>>>(bufA, wpad, bufB, B, 256, 48);
        fc3_logsoftmax_kernel<<<B, 64, 0, stream>>>(bufB, fc3b, (float*)d_out);
    }
}
